// Hybrid_Model_75600014344257
// MI455X (gfx1250) — compile-verified, matrix operand order checked by reading, not run on hardware
//
#include <hip/hip_runtime.h>
#include <cstdint>
#include <cstddef>

// ---------------------------------------------------------------------------
// Hybrid GNN (PAMNet + cross-attn head) forward for MI455X / gfx1250.
// Dense GEMMs -> v_wmma_f32_16x16x32_bf16 (wave32 WMMA), B operand staged in
// LDS via global_load_async_to_lds_b128 (ASYNCcnt) when available.
// Gather/scatter segment-sums use global_atomic_add_f32. Edge/triplet feature
// tensors stored bf16 to halve HBM traffic (23.3 TB/s bound).
// Attention has sequence length 1 => softmax == 1 => mha(v) = Wo(Wv v+bv)+bo.
// ---------------------------------------------------------------------------

#define DEV __device__ __forceinline__

typedef unsigned short u16;
typedef unsigned int   u32;
typedef __attribute__((ext_vector_type(16))) __bf16 v16bf;
typedef __attribute__((ext_vector_type(8)))  float  v8f;
typedef __attribute__((ext_vector_type(4)))  int    v4i;

#if defined(__HIP_DEVICE_COMPILE__)
#if __has_builtin(__builtin_amdgcn_global_load_async_to_lds_b128) && \
    __has_builtin(__builtin_amdgcn_s_wait_asynccnt)
#define HAVE_ASYNC_LDS 1
typedef __attribute__((address_space(1))) v4i as1_v4i;
typedef __attribute__((address_space(3))) v4i as3_v4i;
#endif
#endif

#define N_NODES 30000
#define G_GRAPH 256
#define EL_E    240000
#define EG_E    480000
#define T_TRI   600000
#define T2_TRI  300000
#define DIM     128
#define KR      16
#define KSR     6
#define KSA     7
#define UD      512
#define CUTOFF  5.0f

DEV u16 f2bf(float f) {
  u32 u = __float_as_uint(f);
  u32 r = (u + 0x7FFFu + ((u >> 16) & 1u)) >> 16;   // round-to-nearest-even
  return (u16)r;
}
DEV float bf2f(u16 h) { return __uint_as_float(((u32)h) << 16); }

// --------------------------- elementwise helpers ---------------------------

__global__ void k_zero_f32(float* p, long n) {
  long i = (long)blockIdx.x * blockDim.x + threadIdx.x;
  if (i < n) p[i] = 0.0f;
}

__global__ void k_f32_to_bf16(const float* in, u16* out, long n) {
  long i = (long)blockIdx.x * blockDim.x + threadIdx.x;
  if (i < n) out[i] = f2bf(in[i]);
}

// x = emb[x_raw]  (f32 + bf16 mirror)
__global__ void k_embed(const float* __restrict__ emb, const int* __restrict__ xr,
                        float* __restrict__ x, u16* __restrict__ xbf, int n) {
  int i = blockIdx.x * blockDim.x + threadIdx.x;
  if (i >= n * DIM) return;
  int node = i / DIM, c = i - node * DIM;
  float v = emb[(size_t)xr[node] * DIM + c];
  x[i] = v;
  xbf[i] = f2bf(v);
}

// ------------------------------ feature kernels ----------------------------

__global__ void k_rbf_edges(const float* __restrict__ pos,
                            const int* __restrict__ jidx, const int* __restrict__ iidx,
                            int E, u16* __restrict__ F, float* __restrict__ distOut) {
  int e = blockIdx.x * blockDim.x + threadIdx.x;
  if (e >= E) return;
  int i = iidx[e], j = jidx[e];
  float dx = pos[3 * i + 0] - pos[3 * j + 0];
  float dy = pos[3 * i + 1] - pos[3 * j + 1];
  float dz = pos[3 * i + 2] - pos[3 * j + 2];
  float d = sqrtf(dx * dx + dy * dy + dz * dz);
  if (distOut) distOut[e] = d;
  const float step = CUTOFF / (float)(KR - 1);
  u16* o = F + (size_t)e * 32;
#pragma unroll
  for (int k = 0; k < KR; ++k) {
    float t = d - (float)k * step;
    o[k] = f2bf(__expf(-10.0f * t * t));
  }
#pragma unroll
  for (int k = KR; k < 32; ++k) o[k] = 0;
}

__global__ void k_sbf(const float* __restrict__ pos, const float* __restrict__ dist_l,
                      const int* __restrict__ ii, const int* __restrict__ jj,
                      const int* __restrict__ kk, const int* __restrict__ eidx,
                      int T, u16* __restrict__ F) {
  int t = blockIdx.x * blockDim.x + threadIdx.x;
  if (t >= T) return;
  int i = ii[t], j = jj[t], k = kk[t];
  float ax = pos[3 * j + 0] - pos[3 * i + 0];
  float ay = pos[3 * j + 1] - pos[3 * i + 1];
  float az = pos[3 * j + 2] - pos[3 * i + 2];
  float bx = pos[3 * k + 0] - pos[3 * j + 0];
  float by = pos[3 * k + 1] - pos[3 * j + 1];
  float bz = pos[3 * k + 2] - pos[3 * j + 2];
  float adotb = ax * bx + ay * by + az * bz;
  float cx = ay * bz - az * by, cy = az * bx - ax * bz, cz = ax * by - ay * bx;
  float cr = sqrtf(cx * cx + cy * cy + cz * cz);
  float ang = atan2f(cr, adotb);
  float d = dist_l[eidx[t]];
  float rad[KSR];
  const float step = CUTOFF / (float)(KSR - 1);   // = 1.0
#pragma unroll
  for (int r = 0; r < KSR; ++r) {
    float u = d - (float)r * step;
    rad[r] = __expf(-10.0f * u * u);
  }
  u16* o = F + (size_t)t * 64;
#pragma unroll
  for (int a = 0; a < KSA; ++a) {
    float ca = __cosf(ang * (float)a);
#pragma unroll
    for (int r = 0; r < KSR; ++r) o[r * KSA + a] = f2bf(rad[r] * ca);
  }
#pragma unroll
  for (int q = KSR * KSA; q < 64; ++q) o[q] = 0;
}

// ---------------------------- scatter / gather -----------------------------

__global__ void k_gather_mul_scatter(const u16* __restrict__ src, const u16* __restrict__ ea,
                                     const int* __restrict__ sidx, const int* __restrict__ didx,
                                     float* __restrict__ dst, int E) {
  int wid = (blockIdx.x * blockDim.x + threadIdx.x) >> 5;
  int lane = threadIdx.x & 31;
  if (wid >= E) return;
  int s = sidx[wid], d = didx[wid];
  uint2 sv = *(const uint2*)(src + (size_t)s * DIM + lane * 4);
  uint2 ev = *(const uint2*)(ea + (size_t)wid * DIM + lane * 4);
  float* dp = dst + (size_t)d * DIM + lane * 4;
  atomicAdd(dp + 0, bf2f(sv.x & 0xFFFF) * bf2f(ev.x & 0xFFFF));
  atomicAdd(dp + 1, bf2f(sv.x >> 16)    * bf2f(ev.x >> 16));
  atomicAdd(dp + 2, bf2f(sv.y & 0xFFFF) * bf2f(ev.y & 0xFFFF));
  atomicAdd(dp + 3, bf2f(sv.y >> 16)    * bf2f(ev.y >> 16));
}

__global__ void k_edge_mul(const u16* __restrict__ xbf, const u16* __restrict__ ea,
                           const int* __restrict__ jidx, int E,
                           u16* __restrict__ h, float* __restrict__ hplus) {
  int wid = (blockIdx.x * blockDim.x + threadIdx.x) >> 5;
  int lane = threadIdx.x & 31;
  if (wid >= E) return;
  int j = jidx[wid];
  uint2 sv = *(const uint2*)(xbf + (size_t)j * DIM + lane * 4);
  uint2 ev = *(const uint2*)(ea + (size_t)wid * DIM + lane * 4);
  float v0 = bf2f(sv.x & 0xFFFF) * bf2f(ev.x & 0xFFFF);
  float v1 = bf2f(sv.x >> 16)    * bf2f(ev.x >> 16);
  float v2 = bf2f(sv.y & 0xFFFF) * bf2f(ev.y & 0xFFFF);
  float v3 = bf2f(sv.y >> 16)    * bf2f(ev.y >> 16);
  size_t off = (size_t)wid * DIM + lane * 4;
  uint2 hv;
  hv.x = (u32)f2bf(v0) | ((u32)f2bf(v1) << 16);
  hv.y = (u32)f2bf(v2) | ((u32)f2bf(v3) << 16);
  *(uint2*)(h + off) = hv;
  *(float4*)(hplus + off) = make_float4(v0, v1, v2, v3);
}

__global__ void k_scatter_add(const float* __restrict__ src, const int* __restrict__ idx,
                              float* __restrict__ dst, int E) {
  int wid = (blockIdx.x * blockDim.x + threadIdx.x) >> 5;
  int lane = threadIdx.x & 31;
  if (wid >= E) return;
  int d = idx[wid];
  float4 v = *(const float4*)(src + (size_t)wid * DIM + lane * 4);
  float* dp = dst + (size_t)d * DIM + lane * 4;
  atomicAdd(dp + 0, v.x);
  atomicAdd(dp + 1, v.y);
  atomicAdd(dp + 2, v.z);
  atomicAdd(dp + 3, v.w);
}

// --------------------------- WMMA GEMM (bf16) ------------------------------
// Packs f32 W[K x N] into B-fragment-major bf16 tiles (32x16 per tile,
// 32 lanes x 16 elems = 512 elems = 1KB per tile).
__global__ void k_pack_b(const float* __restrict__ W, int K, int Kpad, int N,
                         u16* __restrict__ out) {
  int tid = blockIdx.x * blockDim.x + threadIdx.x;
  int total = Kpad * N;
  if (tid >= total) return;
  int NT = N >> 4;
  int e = tid & 15;
  int lane = (tid >> 4) & 31;
  int rest = tid >> 9;
  int nt = rest % NT;
  int kt = rest / NT;
  int v = e >> 1, p = e & 1;
  int lh = lane >> 4;
  int n = (nt << 4) + (lane & 15);
  int k = (v < 4) ? ((kt << 5) + lh * 8 + v * 2 + p)
                  : ((kt << 5) + 16 + lh * 8 + (v - 4) * 2 + p);
  float val = (k < K) ? W[(size_t)k * N + n] : 0.0f;
  out[tid] = f2bf(val);
}

// Inner K-loop over one slab of 8 column tiles. Fully unrolled, no guards.
template <bool LDSB>
DEV void gemm_pass(const u16* __restrict__ Abase, const u16* __restrict__ Bbase,
                   int KT, int NT, int nb, int lane, v8f (&acc)[8]) {
  for (int kt = 0; kt < KT; ++kt) {
    union { v16bf v; uint4 q[2]; } af;
    const u16* ap = Abase + kt * 32;
    af.q[0] = *(const uint4*)ap;            // K 0..7   (hi lanes: 8..15)
    af.q[1] = *(const uint4*)(ap + 16);     // K 16..23 (hi lanes: 24..31)
#pragma unroll
    for (int j = 0; j < 8; ++j) {
      union { v16bf v; uint4 q[2]; } bm;
      const u16* bp;
      if (LDSB) bp = Bbase + ((((kt << 3) + j) << 9) + (lane << 4));
      else      bp = Bbase + (((size_t)(kt * NT + nb + j)) << 9) + (lane << 4);
      bm.q[0] = *(const uint4*)bp;
      bm.q[1] = *(const uint4*)(bp + 8);
      acc[j] = __builtin_amdgcn_wmma_f32_16x16x32_bf16(
          false, af.v, false, bm.v, (short)0, acc[j], false, false);
    }
  }
}

DEV void gemm_epilogue(v8f (&acc)[8], const float* __restrict__ bias, int nb,
                       int lane, int mt, int N, int mode,
                       float* __restrict__ outF, u16* __restrict__ outB) {
#pragma unroll
  for (int j = 0; j < 8; ++j) {
    int col = ((nb + j) << 4) + (lane & 15);
    float bv = bias ? bias[col] : 0.0f;
#pragma unroll
    for (int v = 0; v < 8; ++v) {
      int row = (mt << 4) + v + ((lane >> 4) << 3);
      size_t off = (size_t)row * N + col;
      float val = acc[j][v] + bv;
      if (mode == 1) val = fmaxf(val, 0.0f);
      if (mode == 2) {
        float s = val / (1.0f + __expf(-val));   // SiLU
        float nv = outF[off] + s;
        outF[off] = nv;
        outB[off] = f2bf(nv);
      } else {
        if (outF) outF[off] = val;
        if (outB) outB[off] = f2bf(val);
      }
    }
  }
}

// C[MxN] = A[MxK](bf16 row-major) @ Bpk (+bias, fused epilogue).
// mode 0: linear; 1: relu; 2: residual-SiLU in-place (outF += silu, outB bf16).
// useLds: stage current 8-tile B slab (K*256 bytes) in LDS via async DMA.
// NT must be a multiple of 8 (all N here are 128 or 256).
__global__ void __launch_bounds__(128)
k_gemm_wmma(const u16* __restrict__ A, const u16* __restrict__ Bpk,
            const float* __restrict__ bias, int M, int K, int N, int mode,
            float* __restrict__ outF, u16* __restrict__ outB, int useLds) {
  extern __shared__ u16 ldsB[];
  int tid = threadIdx.x, lane = tid & 31;
  int mt = blockIdx.x * 4 + (tid >> 5);
  int mTiles = M >> 4, KT = K >> 5, NT = N >> 4;
  bool active = mt < mTiles;                 // wave-uniform; no early return so
  int rowA = active ? ((mt << 4) + (lane & 15)) : 0;   // barriers stay uniform
  const u16* Abase = A + (size_t)rowA * K + ((lane >> 4) << 3);

  for (int nb = 0; nb < NT; nb += 8) {
    if (useLds) {
      __syncthreads();                       // previous slab fully consumed
      int chunks = KT << 9;                  // KT*8 tiles * 64 x 16B chunks
      for (int c = tid; c < chunks; c += 128) {
        int t = c >> 6, w = c & 63;
        int kt = t >> 3, nt = nb + (t & 7);
        const u16* src = Bpk + (((size_t)(kt * NT + nt)) << 9) + (w << 3);
        u16* dst = ldsB + ((t << 9) + (w << 3));
#ifdef HAVE_ASYNC_LDS
        __builtin_amdgcn_global_load_async_to_lds_b128(
            (as1_v4i*)(const_cast<u16*>(src)), (as3_v4i*)dst, 0, 0);
#else
        *(uint4*)dst = *(const uint4*)src;
#endif
      }
#ifdef HAVE_ASYNC_LDS
      __builtin_amdgcn_s_wait_asynccnt(0);
#endif
      __syncthreads();                       // slab visible to all waves
    }
    if (active) {
      v8f acc[8];
#pragma unroll
      for (int j = 0; j < 8; ++j)
#pragma unroll
        for (int v = 0; v < 8; ++v) acc[j][v] = 0.0f;
      if (useLds) gemm_pass<true>(Abase, ldsB, KT, NT, nb, lane, acc);
      else        gemm_pass<false>(Abase, Bpk, KT, NT, nb, lane, acc);
      gemm_epilogue(acc, bias, nb, lane, mt, N, mode, outF, outB);
    }
  }
}

// ------------------------------- LayerNorm ---------------------------------
__global__ void k_layernorm(const float* __restrict__ resid, const float* __restrict__ delta,
                            const float* __restrict__ g, const float* __restrict__ b,
                            int D, float* __restrict__ outF, u16* __restrict__ outB,
                            int outStride, int colOff) {
  extern __shared__ float sh[];
  int row = blockIdx.x, t = threadIdx.x;
  float x = resid[(size_t)row * D + t] + delta[(size_t)row * D + t];
  sh[t] = x;
  __syncthreads();
  for (int s = D >> 1; s > 0; s >>= 1) { if (t < s) sh[t] += sh[t + s]; __syncthreads(); }
  float m = sh[0] / (float)D;
  __syncthreads();
  float dv = x - m;
  sh[t] = dv * dv;
  __syncthreads();
  for (int s = D >> 1; s > 0; s >>= 1) { if (t < s) sh[t] += sh[t + s]; __syncthreads(); }
  float var = sh[0] / (float)D;
  float y = g[t] * dv * rsqrtf(var + 1e-5f) + b[t];
  size_t off = (size_t)row * outStride + colOff + t;
  outF[off] = y;
  outB[off] = f2bf(y);
}

__global__ void k_final(const u16* __restrict__ pr1, const float* __restrict__ Wpr2,
                        const float* __restrict__ bpr2, float* __restrict__ out,
                        int rows, int K) {
  int r = blockIdx.x * blockDim.x + threadIdx.x;
  if (r >= rows) return;
  float s = bpr2[0];
  for (int k = 0; k < K; ++k) s += bf2f(pr1[(size_t)r * K + k]) * Wpr2[k];
  out[r] = s;
}

// ------------------------------- host driver -------------------------------

static inline int cdiv(long a, long b) { return (int)((a + b - 1) / b); }

extern "C" void kernel_launch(void* const* d_in, const int* in_sizes, int n_in,
                              void* d_out, int out_size, void* d_ws, size_t ws_size,
                              hipStream_t stream) {
  (void)in_sizes; (void)n_in; (void)out_size; (void)ws_size;

  // ---- inputs (params dict flattened depth-first, insertion order) ----
  const float* pos    = (const float*)d_in[0];
  const float* unimol = (const float*)d_in[1];
  const float* emb    = (const float*)d_in[2];
  const float* W_rbf_l = (const float*)d_in[3];  const float* b_rbf_l = (const float*)d_in[4];
  const float* W_rbf_g = (const float*)d_in[5];  const float* b_rbf_g = (const float*)d_in[6];
  const float* W_sbf1  = (const float*)d_in[7];  const float* b_sbf1  = (const float*)d_in[8];
  const float* W_sbf2  = (const float*)d_in[9];  const float* b_sbf2  = (const float*)d_in[10];
  const float* Wg = (const float*)d_in[11];      const float* bg = (const float*)d_in[12];
  const float* Wl = (const float*)d_in[13];      const float* bl = (const float*)d_in[14];
  const float* Wp = (const float*)d_in[15];      const float* bp = (const float*)d_in[16];
  const float* Wu = (const float*)d_in[17];      const float* bu = (const float*)d_in[18];
  // p2u: 19 Wq,20 bq,21 Wk,22 bk,23 Wv,24 bv,25 Wo,26 bo  (Q/K dead: seqlen 1)
  const float* Wv_p2u = (const float*)d_in[23];  const float* bv_p2u = (const float*)d_in[24];
  const float* Wo_p2u = (const float*)d_in[25];  const float* bo_p2u = (const float*)d_in[26];
  const float* Wv_u2p = (const float*)d_in[31];  const float* bv_u2p = (const float*)d_in[32];
  const float* Wo_u2p = (const float*)d_in[33];  const float* bo_u2p = (const float*)d_in[34];
  const float* g1 = (const float*)d_in[35];      const float* b1 = (const float*)d_in[36];
  const float* g2 = (const float*)d_in[37];      const float* b2 = (const float*)d_in[38];
  const float* g3 = (const float*)d_in[39];      const float* b3 = (const float*)d_in[40];
  const float* Wf1 = (const float*)d_in[41];     const float* bf1 = (const float*)d_in[42];
  const float* Wf2 = (const float*)d_in[43];     const float* bf2 = (const float*)d_in[44];
  const float* Wpr1 = (const float*)d_in[45];    const float* bpr1 = (const float*)d_in[46];
  const float* Wpr2 = (const float*)d_in[47];    const float* bpr2 = (const float*)d_in[48];
  const int* x_raw = (const int*)d_in[49];
  const int* batch = (const int*)d_in[50];
  const int* el = (const int*)d_in[51];          // j_l = el, i_l = el+EL
  const int* eg = (const int*)d_in[52];          // j_g = eg, i_g = eg+EG
  const int* idx_i = (const int*)d_in[53];
  const int* idx_j = (const int*)d_in[54];
  const int* idx_k = (const int*)d_in[55];
  const int* idx_kj = (const int*)d_in[56];
  const int* idx_ji = (const int*)d_in[57];
  const int* idx_i_pair  = (const int*)d_in[58];
  const int* idx_j1_pair = (const int*)d_in[59];
  const int* idx_j2_pair = (const int*)d_in[60];
  const int* idx_jj_pair = (const int*)d_in[61];
  const int* idx_ji_pair = (const int*)d_in[62];
  float* out = (float*)d_out;

  // ---- workspace bump allocator ----
  char* base = (char*)d_ws;
  size_t cur = 0;
  auto alloc = [&](size_t bytes) -> char* {
    char* p = base + cur;
    cur = (cur + bytes + 255) & ~(size_t)255;
    return p;
  };

  u16* ea_rbf_g = (u16*)alloc((size_t)EG_E * DIM * 2);
  u16* ea_rbf_l = (u16*)alloc((size_t)EL_E * DIM * 2);
  u16* ea_sbf1  = (u16*)alloc((size_t)T2_TRI * DIM * 2);
  u16* ea_sbf2  = (u16*)alloc((size_t)T_TRI * DIM * 2);
  float* x   = (float*)alloc((size_t)N_NODES * DIM * 4);
  u16*   xbf = (u16*)alloc((size_t)N_NODES * DIM * 2);
  float* agg   = (float*)alloc((size_t)N_NODES * DIM * 4);
  u16*   aggbf = (u16*)alloc((size_t)N_NODES * DIM * 2);
  float* dist_l = (float*)alloc((size_t)EL_E * 4);

  // Region shared between preprocessing F matrices and per-layer h/hplus
  size_t hBytes = (size_t)EL_E * DIM * 2;
  size_t hpBytes = (size_t)EL_E * DIM * 4;
  size_t fBytes = (size_t)T_TRI * 64 * 2 + (size_t)T2_TRI * 64 * 2
                + (size_t)EL_E * 32 * 2 + (size_t)EG_E * 32 * 2 + 1024;
  size_t regBytes = hBytes + hpBytes; if (fBytes > regBytes) regBytes = fBytes;
  char* region = alloc(regBytes);
  u16* h_bf    = (u16*)region;
  float* hplus = (float*)(region + hBytes);
  u16* F_sbf2  = (u16*)region;
  u16* F_sbf1  = (u16*)(region + (size_t)T_TRI * 64 * 2);
  u16* F_rbf_l = (u16*)(region + (size_t)T_TRI * 64 * 2 + (size_t)T2_TRI * 64 * 2);
  u16* F_rbf_g = (u16*)(region + (size_t)T_TRI * 64 * 2 + (size_t)T2_TRI * 64 * 2
                               + (size_t)EL_E * 32 * 2);

  // tail buffers (G=256 rows)
  float* pamnet = (float*)alloc(G_GRAPH * DIM * 4);
  u16*   pambf  = (u16*)alloc(G_GRAPH * DIM * 2);
  u16*   unibf  = (u16*)alloc(G_GRAPH * UD * 2);
  float* pp     = (float*)alloc(G_GRAPH * DIM * 4);
  u16*   ppbf   = (u16*)alloc(G_GRAPH * DIM * 2);
  float* up     = (float*)alloc(G_GRAPH * DIM * 4);
  u16*   upbf   = (u16*)alloc(G_GRAPH * DIM * 2);
  u16*   vbf    = (u16*)alloc(G_GRAPH * DIM * 2);
  float* attn   = (float*)alloc(G_GRAPH * DIM * 4);
  float* fused  = (float*)alloc(G_GRAPH * 2 * DIM * 4);
  u16*   fusedbf= (u16*)alloc(G_GRAPH * 2 * DIM * 2);
  u16*   ffn1bf = (u16*)alloc(G_GRAPH * 2 * DIM * 2);
  float* ffn2   = (float*)alloc(G_GRAPH * 2 * DIM * 4);
  float* fused2 = (float*)alloc(G_GRAPH * 2 * DIM * 4);
  u16*   fused2bf=(u16*)alloc(G_GRAPH * 2 * DIM * 2);
  u16*   pr1bf  = (u16*)alloc(G_GRAPH * DIM * 2);

  // packed bf16 weights
  u16* pk_rbf_l = (u16*)alloc(32 * DIM * 2);
  u16* pk_rbf_g = (u16*)alloc(32 * DIM * 2);
  u16* pk_sbf1  = (u16*)alloc(64 * DIM * 2);
  u16* pk_sbf2  = (u16*)alloc(64 * DIM * 2);
  u16* pk_Wg    = (u16*)alloc(2 * DIM * DIM * 2);
  u16* pk_Wl    = (u16*)alloc(2 * DIM * DIM * 2);
  u16* pk_Wp    = (u16*)alloc(DIM * DIM * 2);
  u16* pk_Wu    = (u16*)alloc(UD * DIM * 2);
  u16* pk_Wv_p2u = (u16*)alloc(DIM * DIM * 2);
  u16* pk_Wo_p2u = (u16*)alloc(DIM * DIM * 2);
  u16* pk_Wv_u2p = (u16*)alloc(DIM * DIM * 2);
  u16* pk_Wo_u2p = (u16*)alloc(DIM * DIM * 2);
  u16* pk_Wf1   = (u16*)alloc(256 * 256 * 2);
  u16* pk_Wf2   = (u16*)alloc(256 * 256 * 2);
  u16* pk_Wpr1  = (u16*)alloc(256 * DIM * 2);

  const int TB = 256;
  auto packB = [&](const float* W, int K, int Kpad, int N, u16* o) {
    k_pack_b<<<cdiv((long)Kpad * N, TB), TB, 0, stream>>>(W, K, Kpad, N, o);
  };
  auto gemm = [&](const u16* A, const u16* Bpk, const float* bias, int M, int K,
                  int N, int mode, float* oF, u16* oB) {
    int useLds = (K <= 256) ? 1 : 0;               // slab = K*256 bytes <= 64KB
    size_t shmem = useLds ? (size_t)K * 256 : 0;
    k_gemm_wmma<<<cdiv(M / 16, 4), 128, shmem, stream>>>(A, Bpk, bias, M, K, N,
                                                         mode, oF, oB, useLds);
  };

  // ---- weight packing ----
  packB(W_rbf_l, KR, 32, DIM, pk_rbf_l);
  packB(W_rbf_g, KR, 32, DIM, pk_rbf_g);
  packB(W_sbf1, KSR * KSA, 64, DIM, pk_sbf1);
  packB(W_sbf2, KSR * KSA, 64, DIM, pk_sbf2);
  packB(Wg, DIM, DIM, DIM, pk_Wg);
  packB(Wg + DIM * DIM, DIM, DIM, DIM, pk_Wg + DIM * DIM);
  packB(Wl, DIM, DIM, DIM, pk_Wl);
  packB(Wl + DIM * DIM, DIM, DIM, DIM, pk_Wl + DIM * DIM);
  packB(Wp, DIM, DIM, DIM, pk_Wp);
  packB(Wu, UD, UD, DIM, pk_Wu);
  packB(Wv_p2u, DIM, DIM, DIM, pk_Wv_p2u);
  packB(Wo_p2u, DIM, DIM, DIM, pk_Wo_p2u);
  packB(Wv_u2p, DIM, DIM, DIM, pk_Wv_u2p);
  packB(Wo_u2p, DIM, DIM, DIM, pk_Wo_u2p);
  packB(Wf1, 256, 256, 256, pk_Wf1);
  packB(Wf2, 256, 256, 256, pk_Wf2);
  packB(Wpr1, 256, 256, DIM, pk_Wpr1);

  // ---- preprocessing ----
  k_f32_to_bf16<<<cdiv((long)G_GRAPH * UD, TB), TB, 0, stream>>>(unimol, unibf, (long)G_GRAPH * UD);
  k_embed<<<cdiv((long)N_NODES * DIM, TB), TB, 0, stream>>>(emb, x_raw, x, xbf, N_NODES);
  k_rbf_edges<<<cdiv(EL_E, TB), TB, 0, stream>>>(pos, el, el + EL_E, EL_E, F_rbf_l, dist_l);
  k_rbf_edges<<<cdiv(EG_E, TB), TB, 0, stream>>>(pos, eg, eg + EG_E, EG_E, F_rbf_g, nullptr);
  k_sbf<<<cdiv(T_TRI, TB), TB, 0, stream>>>(pos, dist_l, idx_i, idx_j, idx_k, idx_kj, T_TRI, F_sbf2);
  k_sbf<<<cdiv(T2_TRI, TB), TB, 0, stream>>>(pos, dist_l, idx_i_pair, idx_j1_pair, idx_j2_pair,
                                             idx_jj_pair, T2_TRI, F_sbf1);
  gemm(F_rbf_l, pk_rbf_l, b_rbf_l, EL_E, 32, DIM, 0, nullptr, ea_rbf_l);
  gemm(F_rbf_g, pk_rbf_g, b_rbf_g, EG_E, 32, DIM, 0, nullptr, ea_rbf_g);
  gemm(F_sbf1, pk_sbf1, b_sbf1, T2_TRI, 64, DIM, 0, nullptr, ea_sbf1);
  gemm(F_sbf2, pk_sbf2, b_sbf2, T_TRI, 64, DIM, 0, nullptr, ea_sbf2);

  // ---- message passing layers ----
  const long nodeElems = (long)N_NODES * DIM;
  for (int l = 0; l < 2; ++l) {
    k_zero_f32<<<cdiv(nodeElems, TB), TB, 0, stream>>>(agg, nodeElems);
    k_gather_mul_scatter<<<cdiv((long)EG_E * 32, TB), TB, 0, stream>>>(
        xbf, ea_rbf_g, eg, eg + EG_E, agg, EG_E);
    k_f32_to_bf16<<<cdiv(nodeElems, TB), TB, 0, stream>>>(agg, aggbf, nodeElems);
    gemm(aggbf, pk_Wg + (size_t)l * DIM * DIM, bg + l * DIM, N_NODES, DIM, DIM, 2, x, xbf);

    k_edge_mul<<<cdiv((long)EL_E * 32, TB), TB, 0, stream>>>(xbf, ea_rbf_l, el, EL_E, h_bf, hplus);
    k_gather_mul_scatter<<<cdiv((long)T_TRI * 32, TB), TB, 0, stream>>>(
        h_bf, ea_sbf2, idx_kj, idx_ji, hplus, T_TRI);
    k_gather_mul_scatter<<<cdiv((long)T2_TRI * 32, TB), TB, 0, stream>>>(
        h_bf, ea_sbf1, idx_jj_pair, idx_ji_pair, hplus, T2_TRI);
    k_zero_f32<<<cdiv(nodeElems, TB), TB, 0, stream>>>(agg, nodeElems);
    k_scatter_add<<<cdiv((long)EL_E * 32, TB), TB, 0, stream>>>(hplus, el + EL_E, agg, EL_E);
    k_f32_to_bf16<<<cdiv(nodeElems, TB), TB, 0, stream>>>(agg, aggbf, nodeElems);
    gemm(aggbf, pk_Wl + (size_t)l * DIM * DIM, bl + l * DIM, N_NODES, DIM, DIM, 2, x, xbf);
  }

  // ---- pooling + fusion head ----
  k_zero_f32<<<cdiv((long)G_GRAPH * DIM, TB), TB, 0, stream>>>(pamnet, (long)G_GRAPH * DIM);
  k_scatter_add<<<cdiv((long)N_NODES * 32, TB), TB, 0, stream>>>(x, batch, pamnet, N_NODES);
  k_f32_to_bf16<<<cdiv((long)G_GRAPH * DIM, TB), TB, 0, stream>>>(pamnet, pambf, (long)G_GRAPH * DIM);

  gemm(pambf, pk_Wp, bp, G_GRAPH, DIM, DIM, 0, pp, ppbf);
  gemm(unibf, pk_Wu, bu, G_GRAPH, UD, DIM, 0, up, upbf);

  // mha with seqlen 1: attn = Wo(Wv v + bv) + bo (softmax of one logit == 1)
  gemm(upbf, pk_Wv_p2u, bv_p2u, G_GRAPH, DIM, DIM, 0, nullptr, vbf);
  gemm(vbf, pk_Wo_p2u, bo_p2u, G_GRAPH, DIM, DIM, 0, attn, nullptr);
  k_layernorm<<<G_GRAPH, DIM, DIM * 4, stream>>>(pp, attn, g1, b1, DIM, fused, fusedbf, 2 * DIM, 0);

  gemm(ppbf, pk_Wv_u2p, bv_u2p, G_GRAPH, DIM, DIM, 0, nullptr, vbf);
  gemm(vbf, pk_Wo_u2p, bo_u2p, G_GRAPH, DIM, DIM, 0, attn, nullptr);
  k_layernorm<<<G_GRAPH, DIM, DIM * 4, stream>>>(up, attn, g2, b2, DIM, fused, fusedbf, 2 * DIM, DIM);

  gemm(fusedbf, pk_Wf1, bf1, G_GRAPH, 256, 256, 1, nullptr, ffn1bf);
  gemm(ffn1bf, pk_Wf2, bf2, G_GRAPH, 256, 256, 0, ffn2, nullptr);
  k_layernorm<<<G_GRAPH, 256, 256 * 4, stream>>>(fused, ffn2, g3, b3, 256, fused2, fused2bf, 256, 0);

  gemm(fused2bf, pk_Wpr1, bpr1, G_GRAPH, 256, DIM, 1, nullptr, pr1bf);
  k_final<<<1, G_GRAPH, 0, stream>>>(pr1bf, Wpr2, bpr2, out, G_GRAPH, DIM);
}
